// ContextAttention_39161511805579
// MI455X (gfx1250) — matrix operand order repaired and checked by reading, not compiled, not run on hardware
//
#include <hip/hip_runtime.h>

// ---------------------------------------------------------------------------
// Contextual attention for MI455X (gfx1250), bf16 WMMA pipeline + TDM staging.
//   f,b: (4,128,96,96) f32   mask: (4,1,96,96) f32
//   out: y (4,128,96,96) f32  ++  offsets (4,2,48,48) i32 (raw bits)
// ---------------------------------------------------------------------------

#define BSZ 4
#define CCH 128
#define HF  96
#define IH  48
#define LL  2304   // 48*48

typedef __attribute__((ext_vector_type(16))) __bf16 bf16x16;
typedef __attribute__((ext_vector_type(8)))  float  f32x8;
typedef unsigned int u32x4 __attribute__((ext_vector_type(4)));
typedef int          i32x8 __attribute__((ext_vector_type(8)));
typedef int          i32x4 __attribute__((ext_vector_type(4)));

#if defined(__AMDGCN__) && __has_builtin(__builtin_amdgcn_tensor_load_to_lds) && \
    __has_builtin(__builtin_amdgcn_s_wait_tensorcnt)
#define USE_TDM 1
#else
#define USE_TDM 0
#endif

struct Chunk32 { float4 lo, hi; };   // 32 bytes = one bf16x16 fragment

__device__ inline f32x8 wmma_bf16(bf16x16 a, bf16x16 b, f32x8 c) {
  // D = A(16x32) x B(32x16) + C, fp32 accumulate
  return __builtin_amdgcn_wmma_f32_16x16x32_bf16(false, a, false, b,
                                                 (short)0, c, false, false);
}

// A fragment: row m0+(lane&15); K chunks [kb..kb+7] and [kb+16..kb+23], kb=8*(lane>>4)
__device__ inline bf16x16 load_a_frag(const __bf16* __restrict__ A, int lda,
                                      int m0, int k0, int lane) {
  int r  = m0 + (lane & 15);
  int kb = k0 + ((lane >> 4) << 3);
  const __bf16* p = A + (size_t)r * lda + kb;
  Chunk32 c;
  c.lo = *reinterpret_cast<const float4*>(p);
  c.hi = *reinterpret_cast<const float4*>(p + 16);
  return __builtin_bit_cast(bf16x16, c);
}

// B fragment: column n0+(lane&15) stored as contiguous K; K chunks [kb..kb+7] and [kb+16..kb+23], kb=8*(lane>>4)
__device__ inline bf16x16 load_b_frag(const __bf16* __restrict__ B, int ldb,
                                      int n0, int k0, int lane) {
  int r  = n0 + (lane & 15);
  int kb = k0 + ((lane >> 4) << 3);
  const __bf16* p = B + (size_t)r * ldb + kb;
  Chunk32 c;
  c.lo = *reinterpret_cast<const float4*>(p);
  c.hi = *reinterpret_cast<const float4*>(p + 16);
  return __builtin_bit_cast(bf16x16, c);
}

// B fragment with explicit per-lane source row + validity (zero-fill when invalid)
__device__ inline bf16x16 load_b_frag_row(const __bf16* __restrict__ B, int ldb,
                                          int row, int k0, int lane, bool valid) {
  Chunk32 c;
  if (valid) {
    const __bf16* p = B + (size_t)row * ldb + k0 + ((lane >> 4) << 3);
    c.lo = *reinterpret_cast<const float4*>(p);
    c.hi = *reinterpret_cast<const float4*>(p + 16);
  } else {
    c.lo = make_float4(0.f, 0.f, 0.f, 0.f);
    c.hi = make_float4(0.f, 0.f, 0.f, 0.f);
  }
  return __builtin_bit_cast(bf16x16, c);
}

__device__ inline f32x8 zero8() {
  f32x8 z; for (int v = 0; v < 8; ++v) z[v] = 0.f; return z;
}

// ---------------------------------------------------------------------------
// Prep: downsample f,b -> (pos,chan) bf16 layouts; per-pos sumsq of bd; mm mask
// ---------------------------------------------------------------------------
__global__ void kprep_ds(const float* __restrict__ f, const float* __restrict__ b,
                         const float* __restrict__ mask,
                         __bf16* __restrict__ fdl, __bf16* __restrict__ bdl,
                         float* __restrict__ sumsq, float* __restrict__ mmb) {
  int idx = blockIdx.x * blockDim.x + threadIdx.x;   // s*LL + a
  if (idx >= BSZ * LL) return;
  int s = idx / LL, a = idx % LL;
  int ay = a / IH, ax = a % IH;
  int sp = (2 * ay) * HF + 2 * ax;
  const float* fs = f + (size_t)s * CCH * HF * HF;
  const float* bs = b + (size_t)s * CCH * HF * HF;
  float ss = 0.f;
  for (int c = 0; c < CCH; ++c) {
    float fv = fs[(size_t)c * HF * HF + sp];
    float bv = bs[(size_t)c * HF * HF + sp];
    fdl[(size_t)idx * CCH + c] = (__bf16)fv;
    bdl[(size_t)idx * CCH + c] = (__bf16)bv;
    ss += bv * bv;
  }
  sumsq[idx] = ss;
  const float* ms = mask + (size_t)s * HF * HF;
  float acc = 0.f;
  for (int dy = -1; dy <= 1; ++dy)
    for (int dx = -1; dx <= 1; ++dx) {
      int yy = ay + dy, xx = ax + dx;
      float mv = (yy >= 0 && yy < IH && xx >= 0 && xx < IH)
                     ? ms[(2 * yy) * HF + 2 * xx] : 0.f;
      acc += 1.f - mv;
    }
  mmb[idx] = (acc * (1.f / 9.f) < 0.85f) ? 1.f : 0.f;
}

// Patch-norm denominator: max(sqrt(sum_{3x3 valid} sumsq + 1e-8), 1e-4)
__global__ void kdenom(const float* __restrict__ sumsq, float* __restrict__ denom) {
  int idx = blockIdx.x * blockDim.x + threadIdx.x;
  if (idx >= BSZ * LL) return;
  int s = idx / LL, a = idx % LL;
  int ay = a / IH, ax = a % IH;
  const float* ssq = sumsq + (size_t)s * LL;
  float n2 = 1e-8f;
  for (int dy = -1; dy <= 1; ++dy)
    for (int dx = -1; dx <= 1; ++dx) {
      int yy = ay + dy, xx = ax + dx;
      if (yy >= 0 && yy < IH && xx >= 0 && xx < IH) n2 += ssq[yy * IH + xx];
    }
  denom[idx] = fmaxf(sqrtf(n2), 1e-4f);
}

// Deconv weights W[tap=u*4+v][c][wi] = raw_w[pi(wi),c,u,v] (bf16), K-permuted
// (pi = 48x48 bi-major swap) to match the permuted K order of Pt.
__global__ void kprep_w(const float* __restrict__ b, int s, __bf16* __restrict__ W) {
  long e = (long)blockIdx.x * blockDim.x + threadIdx.x;   // over 16*128*2304
  if (e >= 16L * CCH * LL) return;
  int tap = (int)(e / (CCH * LL));
  int rem = (int)(e % (CCH * LL));
  int c  = rem / LL;
  int wi = rem % LL;
  int i  = (wi % IH) * IH + wi / IH;                 // logical patch index
  int u = tap >> 2, v = tap & 3;
  long F = 2048L * i + 512L * u + 128L * v + c;      // flat into (ki,kj,c',h2,w2)
  int ki = (int)(F / 1179648L); long r1 = F % 1179648L;
  int kj = (int)(r1 / 294912L); long r2 = r1 % 294912L;
  int cc = (int)(r2 / 2304L);   long r3 = r2 % 2304L;
  int h2 = (int)(r3 / 48L);     int w2 = (int)(r3 % 48L);
  int yy = ki + 2 * h2 - 1, xx = kj + 2 * w2 - 1;    // 1-pad
  float val = (yy >= 0 && yy < HF && xx >= 0 && xx < HF)
                  ? b[(((size_t)s * CCH + cc) * HF + yy) * HF + xx] : 0.f;
  W[((size_t)tap * CCH + c) * LL + wi] = (__bf16)val;
}

// ---------------------------------------------------------------------------
// GEMM 1: Gram matrix  G[a,b] = sum_c bd[a,c] * fd[b,c]   (2304x2304, K=128)
// ---------------------------------------------------------------------------
__global__ __launch_bounds__(256) void kgemm_g(const __bf16* __restrict__ bdl,
                                               const __bf16* __restrict__ fdl,
                                               float* __restrict__ G) {
  int lane = threadIdx.x & 31;
  int wave = threadIdx.x >> 5;
  int wm = wave >> 1, wn = wave & 1;
  int m0 = blockIdx.x * 128 + wm * 32;
  int n0 = blockIdx.y * 128 + wn * 64;

  f32x8 acc[2][4];
  for (int i = 0; i < 2; ++i) for (int j = 0; j < 4; ++j) acc[i][j] = zero8();

  for (int k0 = 0; k0 < CCH; k0 += 32) {
    bf16x16 af[2], bfr[4];
    for (int i = 0; i < 2; ++i) af[i]  = load_a_frag(bdl, CCH, m0 + 16 * i, k0, lane);
    for (int j = 0; j < 4; ++j) bfr[j] = load_b_frag(fdl, CCH, n0 + 16 * j, k0, lane);
    for (int i = 0; i < 2; ++i)
      for (int j = 0; j < 4; ++j)
        acc[i][j] = wmma_bf16(af[i], bfr[j], acc[i][j]);
  }
  int rbase = (lane >> 4) * 8, col = lane & 15;
  for (int i = 0; i < 2; ++i)
    for (int j = 0; j < 4; ++j)
      for (int v = 0; v < 8; ++v)
        G[(size_t)(m0 + 16 * i + rbase + v) * LL + (n0 + 16 * j + col)] = acc[i][j][v];
}

// ---------------------------------------------------------------------------
// raw score: 9-diagonal-tap stencil on G / denom[a].
// Stages 9 row segments (3 groups of 3 consecutive rows, 360 f32 each) into
// LDS with the Tensor Data Mover (wave 0 issues 3 TDM descriptors).
// ---------------------------------------------------------------------------
__global__ __launch_bounds__(256) void kraw(const float* __restrict__ G,
                                            const float* __restrict__ denom,
                                            float* __restrict__ S) {
  __shared__ float tile[9 * 360];          // offset 0 in LDS (only shared array)
  const int a = blockIdx.x;
  const int colbase = blockIdx.y * 256;
  const int c0 = colbase - 52;             // 16B-aligned staging origin

#if USE_TDM
  if (threadIdx.x < 32) {
    for (int gi = 0; gi < 3; ++gi) {
      int row0 = a + ((gi == 0) ? -49 : (gi == 1) ? -1 : 47);
      unsigned long long ga =
          (unsigned long long)(G + (long)row0 * LL + c0);   // byte address
      u32x4 g0; i32x8 g1; i32x4 g2, g3; i32x8 g4;
      g0[0] = 1u;                                   // count=1, user descriptor
      g0[1] = (unsigned)(gi * 3 * 360 * 4);         // lds_addr (bytes)
      g0[2] = (unsigned)(ga & 0xFFFFFFFFull);       // global_addr[31:0]
      g0[3] = (unsigned)((ga >> 32) & 0x01FFFFFFull) | (2u << 30); // type=2
      g1[0] = (2 << 16);                            // data_size = 4 bytes
      g1[1] = 0;                                    // dim0 lo16 = 0 (dim0=1<<20)
      g1[2] = 0x10 | (16 << 16);                    // dim0 hi16, dim1 lo16 = 16
      g1[3] = (360 << 16);                          // tile_dim0 = 360
      g1[4] = 3;                                    // tile_dim1 = 3, tile_dim2=0
      g1[5] = LL;                                   // tensor_dim0_stride = 2304
      g1[6] = 0; g1[7] = 0;
      g2[0] = 0; g2[1] = 0; g2[2] = 0; g2[3] = 0;
      g3[0] = 0; g3[1] = 0; g3[2] = 0; g3[3] = 0;
      g4[0] = 0; g4[1] = 0; g4[2] = 0; g4[3] = 0;
      g4[4] = 0; g4[5] = 0; g4[6] = 0; g4[7] = 0;
      __builtin_amdgcn_tensor_load_to_lds(g0, g1, g2, g3, g4, 0);
    }
    __builtin_amdgcn_s_wait_tensorcnt(0);
  }
#else
  {
    const long total = (long)LL * LL;
    for (int q = threadIdx.x; q < 9 * 360; q += 256) {
      int r = q / 360, cc = q % 360;
      int od = ((r / 3) == 0 ? -49 : (r / 3) == 1 ? -1 : 47) + (r % 3);
      long gg = (long)(a + od) * LL + (c0 + cc);
      tile[q] = (gg >= 0 && gg < total) ? G[gg] : 0.f;
    }
  }
#endif
  __syncthreads();

  int tid = threadIdx.x;
  int bc = colbase + tid;
  int ay = a / IH, ax = a % IH, by = bc / IH, bx = bc % IH;
  float acc = 0.f;
  for (int dy = -1; dy <= 1; ++dy)
    for (int dx = -1; dx <= 1; ++dx) {
      int r = (dy + 1) * 3 + (dx + 1);
      bool v = (unsigned)(ay + dy) < IH && (unsigned)(ax + dx) < IH &&
               (unsigned)(by + dy) < IH && (unsigned)(bx + dx) < IH;
      float t = tile[r * 360 + tid + (48 * dy + dx) + 52];
      acc += v ? t : 0.f;
    }
  S[(size_t)a * LL + bc] = acc / denom[a];
}

// ---------------------------------------------------------------------------
// fuse pass 1 (flat diagonal 3-tap) fused with the bi-major permutation:
// writes D[pi(a)][pi(b)] fully coalesced via a padded-LDS row transpose.
// ---------------------------------------------------------------------------
__global__ __launch_bounds__(256) void kfuse1t(const float* __restrict__ S,
                                               float* __restrict__ D) {
  __shared__ float row[LL + IH];           // padded: idx = b + b/48, stride 49
  int a = blockIdx.x, tid = threadIdx.x;
  for (int k = 0; k < 9; ++k) {            // 2304 = 9*256
    int b = tid + k * 256;
    float acc = 0.f;
    for (int t = -1; t <= 1; ++t) {
      int qa = a + t, qb = b + t;
      if (qa >= 0 && qa < LL && qb >= 0 && qb < LL)
        acc += S[(size_t)qa * LL + qb];
    }
    row[b + b / IH] = acc;
  }
  __syncthreads();
  int wa = (a % IH) * IH + a / IH;
  for (int k = 0; k < 9; ++k) {
    int wb = tid + k * 256;
    int idx = (wb % IH) * (IH + 1) + wb / IH;   // conflict-free stride 49
    D[(size_t)wa * LL + wb] = row[idx];
  }
}

// ---------------------------------------------------------------------------
// fuse pass 2 (native in permuted space: plain flat diagonal 3-tap) fused
// with masked softmax + argmax. Pt written with permuted K order (matches W).
// ---------------------------------------------------------------------------
__global__ __launch_bounds__(256) void kfuse2sm(const float* __restrict__ SP,
                                                const float* __restrict__ mmb,
                                                __bf16* __restrict__ Pt,
                                                int* __restrict__ offs, int s) {
  __shared__ float sv[256];
  __shared__ int   si[256];
  int wa = blockIdx.x, tid = threadIdx.x;
  int a = (wa % IH) * IH + wa / IH;        // logical A index (pi involution)

  float xv[9], mv[9]; int jv[9];
  float lmax = -3.4e38f;
  for (int k = 0; k < 9; ++k) {
    int wj = tid + k * 256;
    float acc = 0.f;
    for (int t = -1; t <= 1; ++t) {
      int qa = wa + t, qb = wj + t;
      if (qa >= 0 && qa < LL && qb >= 0 && qb < LL)
        acc += SP[(size_t)qa * LL + qb];
    }
    int j = (wj % IH) * IH + wj / IH;      // logical B index
    float mmv = mmb[j];
    float x = 10.f * acc * mmv;
    xv[k] = x; mv[k] = mmv; jv[k] = j;
    lmax = fmaxf(lmax, x);
  }
  sv[tid] = lmax; __syncthreads();
  for (int st = 128; st > 0; st >>= 1) {
    if (tid < st) sv[tid] = fmaxf(sv[tid], sv[tid + st]);
    __syncthreads();
  }
  float m = sv[0]; __syncthreads();

  float lsum = 0.f;
  for (int k = 0; k < 9; ++k) { xv[k] = __expf(xv[k] - m); lsum += xv[k]; }
  sv[tid] = lsum; __syncthreads();
  for (int st = 128; st > 0; st >>= 1) {
    if (tid < st) sv[tid] += sv[tid + st];
    __syncthreads();
  }
  float rinv = 1.f / sv[0]; __syncthreads();

  float bv = -1.f; int bi = LL;
  for (int k = 0; k < 9; ++k) {
    int wj = tid + k * 256;
    float p = xv[k] * rinv * mv[k];
    Pt[(size_t)a * LL + wj] = (__bf16)p;
    if (p > bv || (p == bv && jv[k] < bi)) { bv = p; bi = jv[k]; }
  }
  sv[tid] = bv; si[tid] = bi; __syncthreads();
  for (int st = 128; st > 0; st >>= 1) {
    if (tid < st) {
      if (sv[tid + st] > sv[tid] ||
          (sv[tid + st] == sv[tid] && si[tid + st] < si[tid])) {
        sv[tid] = sv[tid + st]; si[tid] = si[tid + st];
      }
    }
    __syncthreads();
  }
  if (tid == 0) {
    int off = si[0];
    int ay = a / IH, ax = a % IH;
    offs[((s * 2 + 0) * IH + ay) * IH + ax] = off / HF - ay;
    offs[((s * 2 + 1) * IH + ay) * IH + ax] = off % HF - ax;
  }
}

// ---------------------------------------------------------------------------
// GEMM 2: reconstruction. Per parity class (pY,pX): out[c, 2oy+pY, 2ox+pX] =
//   0.25 * sum over 4 taps sum_wi W[tap][c][wi] * Pt[apos(oy,ox,tap)][wi]
// ---------------------------------------------------------------------------
__global__ __launch_bounds__(256) void krecon(const __bf16* __restrict__ W,
                                              const __bf16* __restrict__ Pt,
                                              float* __restrict__ out, int s) {
  int lane = threadIdx.x & 31;
  int wave = threadIdx.x >> 5;
  int wm = wave >> 1, wn = wave & 1;
  int par = blockIdx.y;
  int pY = par >> 1, pX = par & 1;
  int m0  = wm * 32;                     // channel base (M = 128 total)
  int n0w = blockIdx.x * 128 + wn * 64;  // output-position base

  f32x8 acc[2][4];
  for (int i = 0; i < 2; ++i) for (int j = 0; j < 4; ++j) acc[i][j] = zero8();

  for (int tky = 0; tky < 2; ++tky) {
    int ky = pY + 2 * tky;
    int dy = pY - 1 + tky;               // ay = oy + dy
    for (int tkx = 0; tkx < 2; ++tkx) {
      int kx = pX + 2 * tkx;
      int dx = pX - 1 + tkx;             // ax = ox + dx
      int tap = (3 - ky) * 4 + (3 - kx);
      const __bf16* Wt = W + (size_t)tap * CCH * LL;
      for (int k0 = 0; k0 < LL; k0 += 32) {
        bf16x16 af[2];
        af[0] = load_a_frag(Wt, LL, m0, k0, lane);
        af[1] = load_a_frag(Wt, LL, m0 + 16, k0, lane);
        for (int j = 0; j < 4; ++j) {
          int n0 = n0w + 16 * j;
          int oy = n0 / IH;              // uniform within 16-tile (16 | 48)
          int oxb = n0 % IH;
          int ayy = oy + dy;
          if (ayy < 0 || ayy >= IH) continue;   // wave-uniform skip
          int ox  = oxb + (lane & 15);
          int axx = ox + dx;
          bool valid = (axx >= 0 && axx < IH);
          bf16x16 bfv = load_b_frag_row(Pt, LL, ayy * IH + axx, k0, lane, valid);
          acc[0][j] = wmma_bf16(af[0], bfv, acc[0][j]);
          acc[1][j] = wmma_bf16(af[1], bfv, acc[1][j]);
        }
      }
    }
  }
  int rbase = (lane >> 4) * 8, col = lane & 15;
  for (int i = 0; i < 2; ++i)
    for (int j = 0; j < 4; ++j) {
      int n = n0w + 16 * j + col;
      int oy = n / IH, ox = n % IH;
      int Y = 2 * oy + pY, X = 2 * ox + pX;
      for (int v = 0; v < 8; ++v) {
        int c = m0 + 16 * i + rbase + v;
        out[(((size_t)s * CCH + c) * HF + Y) * HF + X] = 0.25f * acc[i][j][v];
      }
    }
}

// ---------------------------------------------------------------------------
extern "C" void kernel_launch(void* const* d_in, const int* in_sizes, int n_in,
                              void* d_out, int out_size, void* d_ws, size_t ws_size,
                              hipStream_t stream) {
  (void)in_sizes; (void)n_in; (void)out_size; (void)ws_size;
  const float* f    = (const float*)d_in[0];
  const float* b    = (const float*)d_in[1];
  const float* mask = (const float*)d_in[2];

  float* y    = (float*)d_out;
  int*   offs = (int*)((float*)d_out + (size_t)BSZ * CCH * HF * HF);

  char* ws = (char*)d_ws;
  size_t off = 0;
  auto carve = [&](size_t bytes) {
    void* p = ws + off;
    off = (off + bytes + 255) & ~(size_t)255;
    return p;
  };
  __bf16* fdl    = (__bf16*)carve((size_t)BSZ * LL * CCH * 2);
  __bf16* bdl    = (__bf16*)carve((size_t)BSZ * LL * CCH * 2);
  float*  sumsq  = (float*)carve((size_t)BSZ * LL * 4);
  float*  denomB = (float*)carve((size_t)BSZ * LL * 4);
  float*  mmb    = (float*)carve((size_t)BSZ * LL * 4);
  __bf16* Wp     = (__bf16*)carve((size_t)16 * CCH * LL * 2);   // per-sample reuse
  float*  SA     = (float*)carve((size_t)LL * LL * 4);          // per-sample reuse
  float*  SB     = (float*)carve((size_t)LL * LL * 4);          // per-sample reuse
  __bf16* Pt     = (__bf16*)carve((size_t)LL * LL * 2);         // per-sample reuse

  kprep_ds<<<(BSZ * LL + 255) / 256, 256, 0, stream>>>(f, b, mask, fdl, bdl, sumsq, mmb);
  kdenom<<<(BSZ * LL + 255) / 256, 256, 0, stream>>>(sumsq, denomB);

  for (int s = 0; s < BSZ; ++s) {
    kprep_w<<<(16 * CCH * LL + 255) / 256, 256, 0, stream>>>(b, s, Wp);
    kgemm_g<<<dim3(LL / 128, LL / 128), 256, 0, stream>>>(
        bdl + (size_t)s * LL * CCH, fdl + (size_t)s * LL * CCH, SA);
    kraw<<<dim3(LL, LL / 256), 256, 0, stream>>>(SA, denomB + (size_t)s * LL, SB);
    kfuse1t<<<LL, 256, 0, stream>>>(SB, SA);
    kfuse2sm<<<LL, 256, 0, stream>>>(SA, mmb + (size_t)s * LL, Pt, offs, s);
    krecon<<<dim3(LL / 128, 4), 256, 0, stream>>>(Wp, Pt, y, s);
  }
}
